// TransformerBlock_58394375357206
// MI455X (gfx1250) — matrix operand order repaired and checked by reading, not compiled, not run on hardware
//
#include <hip/hip_runtime.h>
#include <hip/hip_bf16.h>

// ---------------------------------------------------------------------------
// Types matching CDNA5 WMMA operand layouts (wave32)
// ---------------------------------------------------------------------------
typedef __attribute__((ext_vector_type(16))) __bf16 v16bf;
typedef __attribute__((ext_vector_type(8)))  __bf16 bf16x8;
typedef __attribute__((ext_vector_type(8)))  float  v8f;

#define D_MODEL 768
#define N_HEADS 12
#define D_HEAD  64
#define D_FF    3072
#define T_SEQ   4096

__device__ __forceinline__ v8f wmma_bf16(v16bf a, v16bf b, v8f c) {
    // v_wmma_f32_16x16x32_bf16  (D = A[16x32] * B[32x16] + C, f32 accum)
    return __builtin_amdgcn_wmma_f32_16x16x32_bf16(
        /*neg_a=*/false, a, /*neg_b=*/false, b,
        /*c_mod=*/(short)0, c, /*reuse_a=*/false, /*reuse_b=*/false);
}

// Load a 16-element bf16 fragment as two contiguous 16B chunks.
__device__ __forceinline__ v16bf ld_frag(const __bf16* p0, const __bf16* p1) {
    bf16x8 lo = *(const bf16x8*)p0;
    bf16x8 hi = *(const bf16x8*)p1;
    v16bf r;
#pragma unroll
    for (int i = 0; i < 8; ++i) { r[i] = lo[i]; r[i + 8] = hi[i]; }
    return r;
}

// ---------------------------------------------------------------------------
// fp32 -> bf16 conversion (weights)
// ---------------------------------------------------------------------------
__global__ void cvt_bf16_kernel(const float* __restrict__ in,
                                __bf16* __restrict__ out, int n) {
    int i = blockIdx.x * blockDim.x + threadIdx.x;
    if (i < n) out[i] = (__bf16)in[i];
}

// ---------------------------------------------------------------------------
// LayerNorm (fp32 in) -> bf16 out.  One block (256 thr = 8 waves) per row.
// ---------------------------------------------------------------------------
__global__ __launch_bounds__(256)
void ln_bf16_kernel(const float* __restrict__ x, const float* __restrict__ g,
                    const float* __restrict__ b, __bf16* __restrict__ out, int D) {
    int row = blockIdx.x;
    const float* xr = x + (size_t)row * D;
    float s = 0.f, s2 = 0.f;
    for (int i = threadIdx.x; i < D; i += blockDim.x) {
        float v = xr[i]; s += v; s2 += v * v;
    }
#pragma unroll
    for (int off = 16; off >= 1; off >>= 1) {
        s  += __shfl_xor(s,  off, 32);
        s2 += __shfl_xor(s2, off, 32);
    }
    __shared__ float aS[8], aS2[8];
    int wave = threadIdx.x >> 5, lane = threadIdx.x & 31;
    if (lane == 0) { aS[wave] = s; aS2[wave] = s2; }
    __syncthreads();
    if (threadIdx.x == 0) {
        float t = 0.f, t2 = 0.f;
#pragma unroll
        for (int w = 0; w < 8; ++w) { t += aS[w]; t2 += aS2[w]; }
        float mu  = t / (float)D;
        float var = t2 / (float)D - mu * mu;
        aS[0]  = mu;
        aS2[0] = rsqrtf(var + 1e-5f);
    }
    __syncthreads();
    float mu = aS[0], rstd = aS2[0];
    __bf16* orow = out + (size_t)row * D;
    for (int i = threadIdx.x; i < D; i += blockDim.x)
        orow[i] = (__bf16)((xr[i] - mu) * rstd * g[i] + b[i]);
}

// ---------------------------------------------------------------------------
// GEMM: C[M,N] = act( A[M,K](bf16) * W[N,K]^T(bf16) + bias ) + residual
// Block = 128 thr (4 waves). Block tile 128x64; wave tile 32(M) x 64(N)
// (two 16-row subtiles sharing B fragments -> 8 WMMAs per K-step).
// Explicit ping-pong double buffering (two register sets, K-loop unrolled
// by 2) so the software pipeline needs NO register copies: each half-step
// issues the other set's loads, then runs 8 WMMAs on the current set.
// ---------------------------------------------------------------------------
__global__ __launch_bounds__(128)
void gemm_bf16_kernel(const __bf16* __restrict__ A, const __bf16* __restrict__ W,
                      int M, int N, int K,
                      const float* __restrict__ bias,
                      const float* __restrict__ residual,
                      float* __restrict__ outf, __bf16* __restrict__ outb,
                      int act_gelu) {
    int lane = threadIdx.x & 31, wave = threadIdx.x >> 5;
    int half = lane >> 4, l16 = lane & 15;
    int m0 = blockIdx.x * 128 + wave * 32;
    int n0 = blockIdx.y * 64;

    const __bf16* arow0 = A + (size_t)(m0 + l16) * K + 16 * half;
    const __bf16* arow1 = A + (size_t)(m0 + 16 + l16) * K + 16 * half;
    const __bf16* wrow0 = W + (size_t)(n0 +  0 + l16) * K + 16 * half;
    const __bf16* wrow1 = W + (size_t)(n0 + 16 + l16) * K + 16 * half;
    const __bf16* wrow2 = W + (size_t)(n0 + 32 + l16) * K + 16 * half;
    const __bf16* wrow3 = W + (size_t)(n0 + 48 + l16) * K + 16 * half;

    v8f acc[2][4];
#pragma unroll
    for (int s = 0; s < 2; ++s)
#pragma unroll
        for (int f = 0; f < 4; ++f) acc[s][f] = v8f{};

    // two fragment register sets (ping / pong)
    v16bf pA0, pA1, pB0, pB1, pB2, pB3;   // set P
    v16bf qA0, qA1, qB0, qB1, qB2, qB3;   // set Q

    auto load_set = [&](int step, v16bf& A0, v16bf& A1, v16bf& B0, v16bf& B1,
                        v16bf& B2, v16bf& B3) {
        int k1 = step << 5;
        A0 = ld_frag(arow0 + k1, arow0 + k1 + 8);
        A1 = ld_frag(arow1 + k1, arow1 + k1 + 8);
        B0 = ld_frag(wrow0 + k1, wrow0 + k1 + 8);
        B1 = ld_frag(wrow1 + k1, wrow1 + k1 + 8);
        B2 = ld_frag(wrow2 + k1, wrow2 + k1 + 8);
        B3 = ld_frag(wrow3 + k1, wrow3 + k1 + 8);
    };
    auto do_wmma = [&](const v16bf& A0, const v16bf& A1, const v16bf& B0,
                       const v16bf& B1, const v16bf& B2, const v16bf& B3) {
        acc[0][0] = wmma_bf16(A0, B0, acc[0][0]);
        acc[1][0] = wmma_bf16(A1, B0, acc[1][0]);
        acc[0][1] = wmma_bf16(A0, B1, acc[0][1]);
        acc[1][1] = wmma_bf16(A1, B1, acc[1][1]);
        acc[0][2] = wmma_bf16(A0, B2, acc[0][2]);
        acc[1][2] = wmma_bf16(A1, B2, acc[1][2]);
        acc[0][3] = wmma_bf16(A0, B3, acc[0][3]);
        acc[1][3] = wmma_bf16(A1, B3, acc[1][3]);
    };

    int nk = K >> 5;      // 24 or 96 here (even)
    load_set(0, pA0, pA1, pB0, pB1, pB2, pB3);
    int it = 0;
    for (; it + 1 < nk; it += 2) {
        load_set(it + 1, qA0, qA1, qB0, qB1, qB2, qB3);
        if (it + 2 < nk) {   // global_prefetch_b8 two K-steps ahead
            __builtin_prefetch(wrow0 + ((it + 2) << 5), 0, 0);
            __builtin_prefetch(wrow2 + ((it + 2) << 5), 0, 0);
        }
        do_wmma(pA0, pA1, pB0, pB1, pB2, pB3);
        if (it + 2 < nk)
            load_set(it + 2, pA0, pA1, pB0, pB1, pB2, pB3);
        do_wmma(qA0, qA1, qB0, qB1, qB2, qB3);
    }
    if (it < nk)   // odd tail (not taken for K = 768 / 3072)
        do_wmma(pA0, pA1, pB0, pB1, pB2, pB3);

    // C layout: VGPR r -> row (+ r + 8*half), lane l16 -> col
#pragma unroll
    for (int s = 0; s < 2; ++s) {
#pragma unroll
        for (int f = 0; f < 4; ++f) {
#pragma unroll
            for (int r = 0; r < 8; ++r) {
                int row = m0 + s * 16 + r + 8 * half;
                int col = n0 + f * 16 + l16;
                float v = acc[s][f][r];
                if (bias) v += bias[col];
                if (act_gelu) v = 0.5f * v * (1.0f + erff(v * 0.70710678118f));
                size_t idx = (size_t)row * N + col;
                if (residual) v += residual[idx];
                if (outf) outf[idx] = v;
                if (outb) outb[idx] = (__bf16)v;
            }
        }
    }
}

// ---------------------------------------------------------------------------
// Flash attention (non-causal). qkv bf16 [T, 3*D_MODEL]; out bf16 [T, D_MODEL].
// Grid: (T/128, N_HEADS). Block 256 thr (8 waves); wave w owns 16 query rows.
// K/V tiles (32 keys) double-buffered through registers: tile j+1 is loaded
// from global into per-thread registers while tile j's WMMAs run (load target
// IS the use register, so no rotation copies).
// V is stored transposed in LDS so P*V B-fragments are contiguous.
// ---------------------------------------------------------------------------
__global__ __launch_bounds__(256)
void attn_kernel(const __bf16* __restrict__ qkv, __bf16* __restrict__ out) {
    __shared__ __align__(16) __bf16 Klds[32][64];     // [key][feat]
    __shared__ __align__(16) __bf16 Vt[64][32];       // [feat][key]
    __shared__ __align__(16) __bf16 Plds[8][16][32];  // per-wave P tile

    int lane = threadIdx.x & 31, wave = threadIdx.x >> 5;
    int half = lane >> 4, l16 = lane & 15;
    int head = blockIdx.y;
    int q0 = blockIdx.x * 128 + wave * 16;

    // staging chunk for this thread: one 16B chunk of K and of V per tile
    int kr = threadIdx.x >> 3;        // key row 0..31
    int fo = (threadIdx.x & 7) * 8;   // feature offset 0..56
    const __bf16* kbase = qkv + (size_t)kr * (3 * D_MODEL) + D_MODEL + head * D_HEAD + fo;
    const __bf16* vbase = qkv + (size_t)kr * (3 * D_MODEL) + 2 * D_MODEL + head * D_HEAD + fo;
    const size_t tile_stride = (size_t)32 * (3 * D_MODEL);

    // Q fragments (rows q0..q0+15, feature K-dim 64 -> two 16x32 A-frags)
    const __bf16* qrow = qkv + (size_t)(q0 + l16) * (3 * D_MODEL) + head * D_HEAD;
    v16bf qa0 = ld_frag(qrow +      16 * half, qrow +  8 + 16 * half);
    v16bf qa1 = ld_frag(qrow + 32 + 16 * half, qrow + 40 + 16 * half);

    float mi[8], li[8];
#pragma unroll
    for (int r = 0; r < 8; ++r) { mi[r] = -3.0e38f; li[r] = 0.f; }
    v8f o0 = {}, o1 = {}, o2 = {}, o3 = {};

    const int nj = T_SEQ / 32;
    bf16x8 kreg = *(const bf16x8*)kbase;
    bf16x8 vreg = *(const bf16x8*)vbase;

    for (int j = 0; j < nj; ++j) {
        // ---- commit staged tile to LDS (V transposed) ----
        *(bf16x8*)(&Klds[kr][fo]) = kreg;
#pragma unroll
        for (int t = 0; t < 8; ++t) Vt[fo + t][kr] = vreg[t];
        __syncthreads();

        // ---- stage next tile into registers (overlaps compute below) ----
        if (j + 1 < nj) {
            kreg = *(const bf16x8*)(kbase + (size_t)(j + 1) * tile_stride);
            vreg = *(const bf16x8*)(vbase + (size_t)(j + 1) * tile_stride);
        }

        // ---- S[16x32] = Q * K^T  (B[k=feat][n=key]) ----
        v8f s0 = {}, s1 = {};
        {
            v16bf b00 = ld_frag(&Klds[l16][16 * half],      &Klds[l16][16 * half + 8]);
            v16bf b01 = ld_frag(&Klds[l16][32 + 16 * half], &Klds[l16][32 + 16 * half + 8]);
            s0 = wmma_bf16(qa0, b00, s0);
            s0 = wmma_bf16(qa1, b01, s0);
            v16bf b10 = ld_frag(&Klds[16 + l16][16 * half],      &Klds[16 + l16][16 * half + 8]);
            v16bf b11 = ld_frag(&Klds[16 + l16][32 + 16 * half], &Klds[16 + l16][32 + 16 * half + 8]);
            s1 = wmma_bf16(qa0, b10, s1);
            s1 = wmma_bf16(qa1, b11, s1);
        }

        // ---- online softmax (row = r + 8*half, col = lane%16) ----
        float alpha[8];
#pragma unroll
        for (int r = 0; r < 8; ++r) {
            float e0 = s0[r] * 0.125f;   // 1/sqrt(64)
            float e1 = s1[r] * 0.125f;
            float mx = fmaxf(e0, e1);
#pragma unroll
            for (int off = 1; off < 16; off <<= 1)
                mx = fmaxf(mx, __shfl_xor(mx, off, 16));
            float mn = fmaxf(mi[r], mx);
            alpha[r] = __expf(mi[r] - mn);
            float p0 = __expf(e0 - mn);
            float p1 = __expf(e1 - mn);
            float rs = p0 + p1;
#pragma unroll
            for (int off = 1; off < 16; off <<= 1)
                rs += __shfl_xor(rs, off, 16);
            li[r] = li[r] * alpha[r] + rs;
            mi[r] = mn;
            int prow = r + 8 * half;
            Plds[wave][prow][l16]      = (__bf16)p0;
            Plds[wave][prow][16 + l16] = (__bf16)p1;
        }
        // wave-local LDS RAW: DS ops from this wave complete in order
        asm volatile("s_wait_dscnt 0" ::: "memory");

        // ---- rescale O and accumulate P * V ----
#pragma unroll
        for (int r = 0; r < 8; ++r) {
            o0[r] *= alpha[r]; o1[r] *= alpha[r];
            o2[r] *= alpha[r]; o3[r] *= alpha[r];
        }
        v16bf pa = ld_frag(&Plds[wave][l16][16 * half],
                           &Plds[wave][l16][16 * half + 8]);
        v16bf vb0 = ld_frag(&Vt[l16][16 * half],      &Vt[l16][16 * half + 8]);
        v16bf vb1 = ld_frag(&Vt[16 + l16][16 * half], &Vt[16 + l16][16 * half + 8]);
        v16bf vb2 = ld_frag(&Vt[32 + l16][16 * half], &Vt[32 + l16][16 * half + 8]);
        v16bf vb3 = ld_frag(&Vt[48 + l16][16 * half], &Vt[48 + l16][16 * half + 8]);
        o0 = wmma_bf16(pa, vb0, o0);
        o1 = wmma_bf16(pa, vb1, o1);
        o2 = wmma_bf16(pa, vb2, o2);
        o3 = wmma_bf16(pa, vb3, o3);
        __syncthreads();
    }

    // ---- epilogue: divide by l, store bf16 ----
#pragma unroll
    for (int r = 0; r < 8; ++r) {
        float inv = 1.0f / li[r];
        int row = q0 + r + 8 * half;
        __bf16* orow = out + (size_t)row * D_MODEL + head * D_HEAD;
        orow[l16]      = (__bf16)(o0[r] * inv);
        orow[16 + l16] = (__bf16)(o1[r] * inv);
        orow[32 + l16] = (__bf16)(o2[r] * inv);
        orow[48 + l16] = (__bf16)(o3[r] * inv);
    }
}

// ---------------------------------------------------------------------------
// Host-side orchestration
// ---------------------------------------------------------------------------
static inline size_t align256(size_t x) { return (x + 255) & ~(size_t)255; }

extern "C" void kernel_launch(void* const* d_in, const int* in_sizes, int n_in,
                              void* d_out, int out_size, void* d_ws, size_t ws_size,
                              hipStream_t stream) {
    (void)in_sizes; (void)n_in; (void)out_size; (void)ws_size;
    const float* x      = (const float*)d_in[0];
    const float* ln1_g  = (const float*)d_in[1];
    const float* ln1_b  = (const float*)d_in[2];
    const float* qkv_w  = (const float*)d_in[3];
    const float* proj_w = (const float*)d_in[4];
    const float* ln2_g  = (const float*)d_in[5];
    const float* ln2_b  = (const float*)d_in[6];
    const float* fc1_w  = (const float*)d_in[7];
    const float* fc1_b  = (const float*)d_in[8];
    const float* fc2_w  = (const float*)d_in[9];
    const float* fc2_b  = (const float*)d_in[10];
    float* out = (float*)d_out;

    // workspace partition
    char* ws = (char*)d_ws;
    size_t off = 0;
    __bf16* qkvw_b = (__bf16*)(ws + off); off = align256(off + (size_t)3 * D_MODEL * D_MODEL * 2);
    __bf16* projw_b = (__bf16*)(ws + off); off = align256(off + (size_t)D_MODEL * D_MODEL * 2);
    __bf16* fc1w_b  = (__bf16*)(ws + off); off = align256(off + (size_t)D_FF * D_MODEL * 2);
    __bf16* fc2w_b  = (__bf16*)(ws + off); off = align256(off + (size_t)D_MODEL * D_FF * 2);
    __bf16* h1      = (__bf16*)(ws + off); off = align256(off + (size_t)T_SEQ * D_MODEL * 2);
    __bf16* qkv     = (__bf16*)(ws + off); off = align256(off + (size_t)T_SEQ * 3 * D_MODEL * 2);
    __bf16* attno   = (__bf16*)(ws + off); off = align256(off + (size_t)T_SEQ * D_MODEL * 2);
    float*  x2      = (float*) (ws + off); off = align256(off + (size_t)T_SEQ * D_MODEL * 4);
    __bf16* h2      = (__bf16*)(ws + off); off = align256(off + (size_t)T_SEQ * D_MODEL * 2);
    __bf16* mlp     = (__bf16*)(ws + off); off = align256(off + (size_t)T_SEQ * D_FF * 2);

    // 1) weights -> bf16
    {
        int n;
        n = 3 * D_MODEL * D_MODEL;
        cvt_bf16_kernel<<<(n + 255) / 256, 256, 0, stream>>>(qkv_w, qkvw_b, n);
        n = D_MODEL * D_MODEL;
        cvt_bf16_kernel<<<(n + 255) / 256, 256, 0, stream>>>(proj_w, projw_b, n);
        n = D_FF * D_MODEL;
        cvt_bf16_kernel<<<(n + 255) / 256, 256, 0, stream>>>(fc1_w, fc1w_b, n);
        n = D_MODEL * D_FF;
        cvt_bf16_kernel<<<(n + 255) / 256, 256, 0, stream>>>(fc2_w, fc2w_b, n);
    }
    // 2) LN1
    ln_bf16_kernel<<<T_SEQ, 256, 0, stream>>>(x, ln1_g, ln1_b, h1, D_MODEL);
    // 3) qkv = h1 @ qkv_w^T  (bf16 out)
    gemm_bf16_kernel<<<dim3(T_SEQ / 128, (3 * D_MODEL) / 64), 128, 0, stream>>>(
        h1, qkvw_b, T_SEQ, 3 * D_MODEL, D_MODEL,
        nullptr, nullptr, nullptr, qkv, 0);
    // 4) attention
    attn_kernel<<<dim3(T_SEQ / 128, N_HEADS), 256, 0, stream>>>(qkv, attno);
    // 5) x2 = x + attno @ proj_w^T
    gemm_bf16_kernel<<<dim3(T_SEQ / 128, D_MODEL / 64), 128, 0, stream>>>(
        attno, projw_b, T_SEQ, D_MODEL, D_MODEL,
        nullptr, x, x2, nullptr, 0);
    // 6) LN2
    ln_bf16_kernel<<<T_SEQ, 256, 0, stream>>>(x2, ln2_g, ln2_b, h2, D_MODEL);
    // 7) mlp = gelu(h2 @ fc1_w^T + fc1_b)  (bf16 out)
    gemm_bf16_kernel<<<dim3(T_SEQ / 128, D_FF / 64), 128, 0, stream>>>(
        h2, fc1w_b, T_SEQ, D_FF, D_MODEL,
        fc1_b, nullptr, nullptr, mlp, 1);
    // 8) out = x2 + mlp @ fc2_w^T + fc2_b
    gemm_bf16_kernel<<<dim3(T_SEQ / 128, D_MODEL / 64), 128, 0, stream>>>(
        mlp, fc2w_b, T_SEQ, D_MODEL, D_FF,
        fc2_b, x2, out, nullptr, 0);
}
